// SHSA_4054449128127
// MI455X (gfx1250) — matrix operand order repaired and checked by reading, not compiled, not run on hardware
//
#include <hip/hip_runtime.h>

#define CDIM  448
#define PD    96
#define QKC   16
#define OC    128      // 2*QK + PDIM
#define NPIX  1024
#define NBATCH 64
#define EPSV  1e-5f
#define PSTRIDE 1032   // Pbuf row stride in u16: 2064B -> 4-bank skew, 16B aligned

typedef __attribute__((ext_vector_type(16))) __bf16 v16bf;
typedef __attribute__((ext_vector_type(8)))  float  v8f;

union ABf { v16bf v; uint4 q[2]; unsigned int d[8]; unsigned short h[16]; };

__device__ __forceinline__ unsigned short f32_to_bf16(float f) {
    __bf16 h = (__bf16)f;                      // native convert on gfx1250
    unsigned short r;
    __builtin_memcpy(&r, &h, 2);
    return r;
}

// A operand: lane's row = two 16B chunks at K = 8hi and K = 16+8hi
__device__ __forceinline__ void loadA(ABf& a, const unsigned short* row, int hi) {
    a.q[0] = *(const uint4*)(row + 8 * hi);
    a.q[1] = *(const uint4*)(row + 16 + 8 * hi);
}
// B operand: lane's column = two 16B chunks at K = 8hi and K = 16+8hi
__device__ __forceinline__ void loadB(ABf& b, const unsigned short* row, int hi) {
    b.q[0] = *(const uint4*)(row + 8 * hi);
    b.q[1] = *(const uint4*)(row + 16 + 8 * hi);
}

// ---------------- GroupNorm stats: one block per batch ----------------
__global__ void __launch_bounds__(256) k_gn_stats(const float* __restrict__ x,
                                                  float* __restrict__ mu,
                                                  float* __restrict__ rstd) {
    int b = blockIdx.x;
    const float* p = x + (size_t)b * CDIM * NPIX;     // x1 = first 96*1024 contiguous floats
    float s1 = 0.f, s2 = 0.f;
    for (int i = threadIdx.x; i < PD * NPIX; i += 256) { float v = p[i]; s1 += v; s2 += v * v; }
    __shared__ float r1[256], r2[256];
    r1[threadIdx.x] = s1; r2[threadIdx.x] = s2; __syncthreads();
    for (int off = 128; off > 0; off >>= 1) {
        if (threadIdx.x < off) { r1[threadIdx.x] += r1[threadIdx.x + off]; r2[threadIdx.x] += r2[threadIdx.x + off]; }
        __syncthreads();
    }
    if (threadIdx.x == 0) {
        float inv = 1.0f / (float)(PD * NPIX);
        float m = r1[0] * inv;
        float var = r2[0] * inv - m * m;
        mu[b] = m; rstd[b] = rsqrtf(var + EPSV);
    }
}

// ---------------- Weight folding ----------------
__global__ void __launch_bounds__(256) k_prep(const float* gn_w, const float* gn_b,
        const float* qkv_w, const float* qbn_w, const float* qbn_b, const float* qbn_rm, const float* qbn_rv,
        const float* proj_w, const float* pbn_w, const float* pbn_b, const float* pbn_rm, const float* pbn_rv,
        unsigned short* Wq, float* bias0, float* rowsum,
        unsigned short* W2, float* bias2) {
    int gtid = blockIdx.x * 256 + threadIdx.x;
    if (blockIdx.x == 0 && threadIdx.x < OC) {
        int o = threadIdx.x;
        float sc = qbn_w[o] * rsqrtf(qbn_rv[o] + EPSV);
        float rs = 0.f, bb = 0.f;
        for (int c = 0; c < PD; ++c) {
            float w = sc * qkv_w[o * PD + c] * gn_w[c];
            Wq[o * PD + c] = f32_to_bf16(w);
            rs += w;
            bb += qkv_w[o * PD + c] * gn_b[c];
        }
        rowsum[o] = rs;
        bias0[o] = sc * (bb - qbn_rm[o]) + qbn_b[o];
    }
    for (int i = gtid; i < CDIM * CDIM; i += gridDim.x * 256) {
        int o = i / CDIM;
        float sc = pbn_w[o] * rsqrtf(pbn_rv[o] + EPSV);
        W2[i] = f32_to_bf16(sc * proj_w[i]);
    }
    if (gtid < CDIM) {
        float sc = pbn_w[gtid] * rsqrtf(pbn_rv[gtid] + EPSV);
        bias2[gtid] = pbn_b[gtid] - sc * pbn_rm[gtid];
    }
}

// ---------------- Transpose + bf16 convert: x1 -> x1bfT[b][n][96], relu(x2) -> yT[b][n][448] c>=96 ----
__global__ void __launch_bounds__(256) k_convert(const float* __restrict__ x,
        unsigned short* __restrict__ x1bfT, unsigned short* __restrict__ yT) {
    int blk = blockIdx.x;
    int nt = blk & 15;          // 16 tiles of 64 pixels
    int ct = (blk >> 4) % 14;   // 14 tiles of 32 channels
    int b  = blk / 224;
    int c = ct * 32 + (threadIdx.x & 31);
    int nbase = nt * 64 + (threadIdx.x >> 5) * 8;
    const float* xp = x + ((size_t)b * CDIM + c) * NPIX + nbase;
    #pragma unroll
    for (int i = 0; i < 8; ++i) {
        int n = nbase + i;
        float f = xp[i];
        if (c < PD) x1bfT[((size_t)b * NPIX + n) * PD + c] = f32_to_bf16(f);
        else        yT[((size_t)b * NPIX + n) * CDIM + c] = f32_to_bf16(fmaxf(f, 0.f));
    }
}

// ---------------- QKV GEMM: 128x1024 per batch, K=96; each wave -> 32x64 output tile ----------------
__global__ void __launch_bounds__(256) k_qkv(const unsigned short* __restrict__ x1bfT,
        const unsigned short* __restrict__ Wq,
        const float* __restrict__ mu, const float* __restrict__ rstd,
        const float* __restrict__ bias0, const float* __restrict__ rowsum,
        unsigned short* __restrict__ qT, unsigned short* __restrict__ kT,
        unsigned short* __restrict__ vbuf) {
    int wg  = blockIdx.x * 8 + (threadIdx.x >> 5);
    int lid = threadIdx.x & 31, hi = lid >> 4, col = lid & 15;
    int ns = wg & 15, otp = (wg >> 4) & 3, b = wg >> 6;
    int n0 = ns * 64;

    v8f acc[2][4];
    #pragma unroll
    for (int u = 0; u < 2; ++u)
        #pragma unroll
        for (int t = 0; t < 4; ++t)
            #pragma unroll
            for (int j = 0; j < 8; ++j) acc[u][t][j] = 0.f;

    int ro = otp * 32 + col;
    for (int s = 0; s < 3; ++s) {
        int c0 = s * 32;
        ABf a0, a1;
        loadA(a0, Wq + ro * PD + c0, hi);
        loadA(a1, Wq + (ro + 16) * PD + c0, hi);
        #pragma unroll
        for (int t = 0; t < 4; ++t) {
            int n = n0 + t * 16 + col;
            ABf bb;
            loadB(bb, x1bfT + ((size_t)b * NPIX + n) * PD + c0, hi);
            acc[0][t] = __builtin_amdgcn_wmma_f32_16x16x32_bf16(false, a0.v, false, bb.v,
                                                                (short)0, acc[0][t], false, false);
            acc[1][t] = __builtin_amdgcn_wmma_f32_16x16x32_bf16(false, a1.v, false, bb.v,
                                                                (short)0, acc[1][t], false, false);
        }
    }
    float mub = mu[b], rstdb = rstd[b];
    #pragma unroll
    for (int u = 0; u < 2; ++u) {
        int ot = otp * 2 + u;
        #pragma unroll
        for (int t = 0; t < 4; ++t) {
            int n = n0 + t * 16 + col;
            union { unsigned short h[8]; uint4 q; } pk;
            #pragma unroll
            for (int j = 0; j < 8; ++j) {
                int o = ot * 16 + j + 8 * hi;
                float val = rstdb * (acc[u][t][j] - mub * rowsum[o]) + bias0[o];
                pk.h[j] = f32_to_bf16(val);
            }
            if (ot == 0)      *(uint4*)(qT + ((size_t)b * NPIX + n) * 32 + 8 * hi) = pk.q;
            else if (ot == 1) *(uint4*)(kT + ((size_t)b * NPIX + n) * 32 + 8 * hi) = pk.q;
            else {
                #pragma unroll
                for (int j = 0; j < 8; ++j) {
                    int o = ot * 16 + j + 8 * hi;
                    vbuf[((size_t)(b * PD + (o - 2 * QKC))) * NPIX + n] = pk.h[j];
                }
            }
        }
    }
}

// ---------------- Flash-style attention: one block per (batch, 16-query tile) ----------------
__global__ void __launch_bounds__(256) k_attn(const unsigned short* __restrict__ qT,
        const unsigned short* __restrict__ kT,
        const unsigned short* __restrict__ vbuf,
        unsigned short* __restrict__ yT) {
    __shared__ __attribute__((aligned(16))) unsigned short Pbuf[16 * PSTRIDE]; // exp(logits), [n][m]
    __shared__ float partial[8][16];
    __shared__ float rowInv[16];

    int b = blockIdx.x >> 6, nt = blockIdx.x & 63;
    int w = threadIdx.x >> 5;
    int lid = threadIdx.x & 31, hi = lid >> 4, col = lid & 15;
    int n0 = nt * 16;

    // A operand: q^T tile (rows = queries, K = qk channel, zero-padded to 32)
    ABf aq;
    loadA(aq, qT + ((size_t)b * NPIX + (n0 + col)) * 32, hi);

    float rs[8];
    #pragma unroll
    for (int j = 0; j < 8; ++j) rs[j] = 0.f;

    for (int i = 0; i < 8; ++i) {                 // each wave handles 8 of 64 m-tiles
        int mt = w + 8 * i;
        int m = mt * 16 + col;
        ABf bk;
        loadB(bk, kT + ((size_t)b * NPIX + m) * 32, hi);
        v8f s;
        #pragma unroll
        for (int j = 0; j < 8; ++j) s[j] = 0.f;
        s = __builtin_amdgcn_wmma_f32_16x16x32_bf16(false, aq.v, false, bk.v,
                                                    (short)0, s, false, false);
        #pragma unroll
        for (int j = 0; j < 8; ++j) {
            float e = __expf(s[j] * 0.25f);       // scale = QK^-0.5; logits O(1) post-BN
            rs[j] += e;
            Pbuf[(j + 8 * hi) * PSTRIDE + m] = f32_to_bf16(e);
        }
    }
    // row-sum: shuffle across the 16 columns, then across 8 waves via LDS (fixed order)
    #pragma unroll
    for (int j = 0; j < 8; ++j) {
        float v = rs[j];
        v += __shfl_xor(v, 1); v += __shfl_xor(v, 2);
        v += __shfl_xor(v, 4); v += __shfl_xor(v, 8);
        rs[j] = v;
    }
    if (col == 0) {
        #pragma unroll
        for (int j = 0; j < 8; ++j) partial[w][8 * hi + j] = rs[j];
    }
    __syncthreads();
    if (threadIdx.x < 16) {
        float t = 0.f;
        #pragma unroll
        for (int ww = 0; ww < 8; ++ww) t += partial[ww][threadIdx.x];
        rowInv[threadIdx.x] = 1.0f / t;
    }
    __syncthreads();

    // Phase 2: x1o(96x16) = V(96x1024) @ P^T; 6 waves x one 16-c tile, K=1024
    if (w < 6) {
        int ct = w;
        v8f acc;
        #pragma unroll
        for (int j = 0; j < 8; ++j) acc[j] = 0.f;
        const unsigned short* vp0 = vbuf + ((size_t)(b * PD + ct * 16 + col)) * NPIX;
        const unsigned short* prow = Pbuf + col * PSTRIDE;
        for (int s = 0; s < 32; ++s) {
            int m0 = s * 32;
            ABf av, bp;
            loadA(av, vp0 + m0, hi);
            loadB(bp, prow + m0, hi);
            acc = __builtin_amdgcn_wmma_f32_16x16x32_bf16(false, av.v, false, bp.v,
                                                          (short)0, acc, false, false);
        }
        float inv = rowInv[col];
        union { unsigned short h[8]; uint4 q; } pk;
        #pragma unroll
        for (int j = 0; j < 8; ++j)
            pk.h[j] = f32_to_bf16(fmaxf(acc[j] * inv, 0.f));   // relu(x1o) pre-applied
        *(uint4*)(yT + ((size_t)b * NPIX + (n0 + col)) * CDIM + ct * 16 + 8 * hi) = pk.q;
    }
}

// ---------------- Projection GEMM: 448x1024 per batch, K=448; each wave -> 32x64 output tile ----
__global__ void __launch_bounds__(256) k_proj(const unsigned short* __restrict__ yT,
        const unsigned short* __restrict__ W2, const float* __restrict__ bias2,
        float* __restrict__ out) {
    int wg  = blockIdx.x * 8 + (threadIdx.x >> 5);
    int lid = threadIdx.x & 31, hi = lid >> 4, col = lid & 15;
    int ns = wg & 15, ot = (wg >> 4) % 14, b = wg / 224;
    int n0 = ns * 64;
    const unsigned short* yb = yT + (size_t)b * NPIX * CDIM;

    v8f acc[2][4];
    #pragma unroll
    for (int u = 0; u < 2; ++u)
        #pragma unroll
        for (int t = 0; t < 4; ++t)
            #pragma unroll
            for (int j = 0; j < 8; ++j) acc[u][t][j] = 0.f;

    int ro = ot * 32 + col;
    for (int s = 0; s < 14; ++s) {
        int c0 = s * 32;
        ABf a0, a1;
        loadA(a0, W2 + ro * CDIM + c0, hi);
        loadA(a1, W2 + (ro + 16) * CDIM + c0, hi);
        #pragma unroll
        for (int t = 0; t < 4; ++t) {
            int n = n0 + t * 16 + col;
            ABf bb;
            loadB(bb, yb + (size_t)n * CDIM + c0, hi);
            acc[0][t] = __builtin_amdgcn_wmma_f32_16x16x32_bf16(false, a0.v, false, bb.v,
                                                                (short)0, acc[0][t], false, false);
            acc[1][t] = __builtin_amdgcn_wmma_f32_16x16x32_bf16(false, a1.v, false, bb.v,
                                                                (short)0, acc[1][t], false, false);
        }
    }
    #pragma unroll
    for (int u = 0; u < 2; ++u)
        #pragma unroll
        for (int t = 0; t < 4; ++t)
            #pragma unroll
            for (int j = 0; j < 8; ++j) {
                int o = ot * 32 + u * 16 + j + 8 * hi;
                int n = n0 + t * 16 + col;
                out[((size_t)(b * CDIM + o)) * NPIX + n] = acc[u][t][j] + bias2[o];
            }
}

extern "C" void kernel_launch(void* const* d_in, const int* in_sizes, int n_in,
                              void* d_out, int out_size, void* d_ws, size_t ws_size,
                              hipStream_t stream) {
    (void)in_sizes; (void)n_in; (void)out_size; (void)ws_size;
    const float* x       = (const float*)d_in[0];
    const float* gn_w    = (const float*)d_in[1];
    const float* gn_b    = (const float*)d_in[2];
    const float* qkv_w   = (const float*)d_in[3];
    const float* qbn_w   = (const float*)d_in[4];
    const float* qbn_b   = (const float*)d_in[5];
    const float* qbn_rm  = (const float*)d_in[6];
    const float* qbn_rv  = (const float*)d_in[7];
    const float* proj_w  = (const float*)d_in[8];
    const float* pbn_w   = (const float*)d_in[9];
    const float* pbn_b   = (const float*)d_in[10];
    const float* pbn_rm  = (const float*)d_in[11];
    const float* pbn_rv  = (const float*)d_in[12];
    float* out = (float*)d_out;

    char* w = (char*)d_ws;
    size_t off = 0;
    auto alloc = [&](size_t bytes) { void* p = w + off; off = (off + bytes + 255) & ~(size_t)255; return p; };

    float* mu      = (float*)alloc(NBATCH * 4);
    float* rstd    = (float*)alloc(NBATCH * 4);
    float* bias0   = (float*)alloc(OC * 4);
    float* rowsum  = (float*)alloc(OC * 4);
    float* bias2   = (float*)alloc(CDIM * 4);
    unsigned short* Wq     = (unsigned short*)alloc((size_t)OC * PD * 2);
    unsigned short* W2     = (unsigned short*)alloc((size_t)CDIM * CDIM * 2);
    unsigned short* qT     = (unsigned short*)alloc((size_t)NBATCH * NPIX * 32 * 2);
    unsigned short* kT     = (unsigned short*)alloc((size_t)NBATCH * NPIX * 32 * 2);
    unsigned short* vbuf   = (unsigned short*)alloc((size_t)NBATCH * PD * NPIX * 2);
    unsigned short* x1bfT  = (unsigned short*)alloc((size_t)NBATCH * NPIX * PD * 2);
    unsigned short* yT     = (unsigned short*)alloc((size_t)NBATCH * NPIX * CDIM * 2);

    // zero the K-padding rows (16..31) of q^T / k^T so the bf16 K=32 WMMA sees zeros
    hipMemsetAsync(qT, 0, (size_t)NBATCH * NPIX * 32 * 2, stream);
    hipMemsetAsync(kT, 0, (size_t)NBATCH * NPIX * 32 * 2, stream);

    k_prep<<<64, 256, 0, stream>>>(gn_w, gn_b, qkv_w, qbn_w, qbn_b, qbn_rm, qbn_rv,
                                   proj_w, pbn_w, pbn_b, pbn_rm, pbn_rv,
                                   Wq, bias0, rowsum, W2, bias2);
    k_gn_stats<<<NBATCH, 256, 0, stream>>>(x, mu, rstd);
    k_convert<<<NBATCH * 14 * 16, 256, 0, stream>>>(x, x1bfT, yT);
    k_qkv<<<512, 256, 0, stream>>>(x1bfT, Wq, mu, rstd, bias0, rowsum, qT, kT, vbuf);
    k_attn<<<NBATCH * 64, 256, 0, stream>>>(qT, kT, vbuf, yT);
    k_proj<<<1792, 256, 0, stream>>>(yT, W2, bias2, out);
}
